// GraphTransformerModel_29308856828501
// MI455X (gfx1250) — matrix operand order repaired and checked by reading, not compiled, not run on hardware
//
#include <hip/hip_runtime.h>
#include <hip/hip_bf16.h>

typedef __attribute__((ext_vector_type(16))) _Float16 v16h;
typedef __attribute__((ext_vector_type(8)))  _Float16 v8h;
typedef __attribute__((ext_vector_type(8)))  float    v8f;

#define GN 100000
#define GE 1600000
#define GH 4
#define GDH 16
#define GHID 64
#define GOC 32
#define GEPS 1e-5f

// ---------------- utility kernels ----------------

__global__ void cvt_f32_to_f16(const float* __restrict__ src, _Float16* __restrict__ dst, int n) {
    int i = blockIdx.x * blockDim.x + threadIdx.x;
    if (i < n) dst[i] = (_Float16)src[i];
}

// Pack Wq/Wk/Wv/Ws (each [3,64,64]) into w16 laid out [l][4][64*64], and Wh [64,32] into wh16.
__global__ void pack_weights(const float* __restrict__ Wq, const float* __restrict__ Wk,
                             const float* __restrict__ Wv, const float* __restrict__ Ws,
                             const float* __restrict__ Wh,
                             _Float16* __restrict__ w16, _Float16* __restrict__ wh16) {
    int i = blockIdx.x * blockDim.x + threadIdx.x;
    if (i < 3 * 4096) {
        int l = i >> 12, r = i & 4095;
        _Float16* base = w16 + (size_t)(l * 4) * 4096;
        base[0 * 4096 + r] = (_Float16)Wq[i];
        base[1 * 4096 + r] = (_Float16)Wk[i];
        base[2 * 4096 + r] = (_Float16)Wv[i];
        base[3 * 4096 + r] = (_Float16)Ws[i];
    }
    if (i < GHID * GOC) wh16[i] = (_Float16)Wh[i];
}

__global__ void init_stats(float* __restrict__ amax, float* __restrict__ denom, int n) {
    int i = blockIdx.x * blockDim.x + threadIdx.x;
    if (i < n) { amax[i] = -3.0e38f; denom[i] = 0.0f; }
}

// ---------------- WMMA fragment helpers ----------------

// A fragment: 16x32 f16 tile of X, rows row0..row0+15, K columns kb..kb+31.
// Lane layout (ISA 7.12.2): lane m = L&15; halves 0..7 = K kk0..kk0+7,
// halves 8..15 = K kk0+16..kk0+23, kk0 = (L<16 ? 0 : 8).
__device__ inline v16h load_a_frag(const _Float16* __restrict__ xh, int row0, int kb, int lane) {
    int m   = lane & 15;
    int kk0 = (lane < 16) ? 0 : 8;
    const _Float16* xr = xh + (size_t)(row0 + m) * GHID + kb + kk0;
    v8h lo = *(const v8h*)(xr);
    v8h hi = *(const v8h*)(xr + 16);
    v16h a;
#pragma unroll
    for (int i = 0; i < 8; ++i) { a[i] = lo[i]; a[8 + i] = hi[i]; }
    return a;
}

// B fragment: 32x16 f16 tile of W (row-major [K][Ncols]); lane supplies N = n0+(L&15),
// halves 0..7 = K kk0..kk0+7, halves 8..15 = K kk0+16..kk0+23, kk0 = (L<16 ? 0 : 8).
__device__ inline v16h load_b_frag(const _Float16* __restrict__ w, int ld, int kb, int n0, int lane) {
    int n   = lane & 15;
    int kk0 = (lane < 16) ? 0 : 8;
    const _Float16* wc = w + (size_t)(kb + kk0) * ld + n0 + n;
    v16h b;
#pragma unroll
    for (int i = 0; i < 8; ++i) { b[i] = wc[(size_t)i * ld]; b[8 + i] = wc[(size_t)(16 + i) * ld]; }
    return b;
}

// ---------------- fused Q/K/V/Skip GEMM (WMMA) ----------------
// block = 128 threads = 4 waves; wave 0->q(f16) 1->k(f16) 2->v(f16) 3->skip(f32).
// Each wave computes a full 16x64 output tile for node rows [blockIdx.x*16, +16).
__global__ void gemm_qkvs(const _Float16* __restrict__ xh, const _Float16* __restrict__ w4,
                          const float* __restrict__ bq, const float* __restrict__ bk,
                          const float* __restrict__ bv, const float* __restrict__ bs,
                          _Float16* __restrict__ qh, _Float16* __restrict__ kh,
                          _Float16* __restrict__ vh, float* __restrict__ skip, int n) {
    int wave = threadIdx.x >> 5;
    int lane = threadIdx.x & 31;
    int row0 = blockIdx.x * 16;
    if (row0 >= n) return;

    const _Float16* wp = w4 + (size_t)wave * 4096;
    const float* bias;
    _Float16* outh = nullptr;
    float* outf = nullptr;
    switch (wave) {
        case 0: bias = bq; outh = qh; break;
        case 1: bias = bk; outh = kh; break;
        case 2: bias = bv; outh = vh; break;
        default: bias = bs; outf = skip; break;
    }

    // WMMAs run with full EXEC (no divergence before this point).
    v16h a0 = load_a_frag(xh, row0, 0, lane);
    v16h a1 = load_a_frag(xh, row0, 32, lane);

    int m  = lane & 15;
    int hi = lane >> 4;           // 0|1 -> M offset 0|8 for C/D
    int rbase = row0 + hi * 8;

#pragma unroll
    for (int ct = 0; ct < 4; ++ct) {
        int n0 = ct * 16;
        v8f acc = {};
        v16h b0 = load_b_frag(wp, GHID, 0, n0, lane);
        acc = __builtin_amdgcn_wmma_f32_16x16x32_f16(false, a0, false, b0, (short)0, acc, false, false);
        v16h b1 = load_b_frag(wp, GHID, 32, n0, lane);
        acc = __builtin_amdgcn_wmma_f32_16x16x32_f16(false, a1, false, b1, (short)0, acc, false, false);

        int col = n0 + m;
        float bb = bias[col];
        size_t base = (size_t)rbase * GHID + col;
        // Wave-uniform select hoisted OUT of the element loop: two exec regions
        // per column tile, stores form clauses.
        if (outh) {
            _Float16* o = outh + base;
#pragma unroll
            for (int r = 0; r < 8; ++r) o[(size_t)r * GHID] = (_Float16)(acc[r] + bb);
        } else {
            float* o = outf + base;
#pragma unroll
            for (int r = 0; r < 8; ++r) o[(size_t)r * GHID] = acc[r] + bb;
        }
    }
}

// ---------------- final head GEMM (WMMA): [N,64] @ [64,32] + bh -> d_out ----------------
// block = 64 threads = 2 waves; wave w handles output columns [w*16, w*16+16).
__global__ void gemm_head(const _Float16* __restrict__ xh, const _Float16* __restrict__ wh16,
                          const float* __restrict__ bh, float* __restrict__ out, int n) {
    int wave = threadIdx.x >> 5;
    int lane = threadIdx.x & 31;
    int row0 = blockIdx.x * 16;
    if (row0 >= n) return;

    v16h a0 = load_a_frag(xh, row0, 0, lane);
    v16h a1 = load_a_frag(xh, row0, 32, lane);

    int n0 = wave * 16;
    v8f acc = {};
    v16h b0 = load_b_frag(wh16, GOC, 0, n0, lane);
    acc = __builtin_amdgcn_wmma_f32_16x16x32_f16(false, a0, false, b0, (short)0, acc, false, false);
    v16h b1 = load_b_frag(wh16, GOC, 32, n0, lane);
    acc = __builtin_amdgcn_wmma_f32_16x16x32_f16(false, a1, false, b1, (short)0, acc, false, false);

    int m = lane & 15;
    int hi = lane >> 4;
    int col = n0 + m;
    float bb = bh[col];
    int rbase = row0 + hi * 8;
#pragma unroll
    for (int r = 0; r < 8; ++r)
        out[(size_t)(rbase + r) * GOC + col] = acc[r] + bb;
}

// ---------------- edge phase ----------------

__device__ inline void atomicMaxF(float* addr, float val) {
    unsigned int* ua = (unsigned int*)addr;
    unsigned int old = __float_as_uint(*addr);
    while (__uint_as_float(old) < val) {
        unsigned int assumed = old;
        old = atomicCAS(ua, assumed, __float_as_uint(val));
        if (old == assumed) break;
    }
}

// one thread per (edge, head): logit = (q[dst]·k[src]) / sqrt(DH), running segment-max
__global__ void edge_logits(const _Float16* __restrict__ qh, const _Float16* __restrict__ kh,
                            const int* __restrict__ src, const int* __restrict__ dst,
                            float* __restrict__ alpha, float* __restrict__ amax, int ne) {
    int idx = blockIdx.x * blockDim.x + threadIdx.x;
    if (idx >= ne * GH) return;
    int e = idx >> 2, h = idx & 3;
    int s = src[e], d = dst[e];
    v16h qv = *(const v16h*)(qh + (size_t)d * GHID + h * GDH);
    v16h kv = *(const v16h*)(kh + (size_t)s * GHID + h * GDH);
    float acc = 0.0f;
#pragma unroll
    for (int i = 0; i < 16; ++i) acc = fmaf((float)qv[i], (float)kv[i], acc);
    acc *= 0.25f;  // 1/sqrt(16)
    alpha[idx] = acc;
    atomicMaxF(amax + (size_t)d * GH + h, acc);
}

// one thread per (edge, head): ex = exp(logit - max); segment-sum into denom
__global__ void edge_exp(float* __restrict__ alpha, const float* __restrict__ amax,
                         const int* __restrict__ dst, float* __restrict__ denom, int ne) {
    int idx = blockIdx.x * blockDim.x + threadIdx.x;
    if (idx >= ne * GH) return;
    int e = idx >> 2, h = idx & 3;
    int d = dst[e];
    float ex = __expf(alpha[idx] - amax[(size_t)d * GH + h]);
    alpha[idx] = ex;
    atomicAdd(denom + (size_t)d * GH + h, ex);
}

// one thread per (edge, channel): accumulate a*v[src] into skip (= x·Ws + bs accumulator)
__global__ void edge_scatter(const _Float16* __restrict__ vh, const float* __restrict__ alpha,
                             const float* __restrict__ denom, const int* __restrict__ src,
                             const int* __restrict__ dst, float* __restrict__ acc, int ne) {
    int idx = blockIdx.x * blockDim.x + threadIdx.x;
    if (idx >= ne * GHID) return;
    int e = idx >> 6, c = idx & 63, h = c >> 4;
    int s = src[e], d = dst[e];
    float w = alpha[(size_t)e * GH + h] / denom[(size_t)d * GH + h];
    float val = (float)vh[(size_t)s * GHID + c] * w;
    atomicAdd(acc + (size_t)d * GHID + c, val);
}

// BatchNorm (eval) + ReLU
__global__ void bn_relu(const float* __restrict__ hsum,
                        const float* __restrict__ g, const float* __restrict__ b,
                        const float* __restrict__ rm, const float* __restrict__ rv,
                        float* __restrict__ xout, int n) {
    int i = blockIdx.x * blockDim.x + threadIdx.x;
    if (i >= n) return;
    int c = i & 63;
    float y = (hsum[i] - rm[c]) * rsqrtf(rv[c] + GEPS) * g[c] + b[c];
    xout[i] = fmaxf(y, 0.0f);
}

// ---------------- launch ----------------

extern "C" void kernel_launch(void* const* d_in, const int* in_sizes, int n_in,
                              void* d_out, int out_size, void* d_ws, size_t ws_size,
                              hipStream_t stream) {
    const float* x   = (const float*)d_in[0];
    const int*   ei  = (const int*)d_in[1];
    const float* Wq  = (const float*)d_in[2];  const float* bq  = (const float*)d_in[3];
    const float* Wk  = (const float*)d_in[4];  const float* bk  = (const float*)d_in[5];
    const float* Wv  = (const float*)d_in[6];  const float* bv  = (const float*)d_in[7];
    const float* Ws  = (const float*)d_in[8];  const float* bs  = (const float*)d_in[9];
    const float* bng = (const float*)d_in[10]; const float* bnb = (const float*)d_in[11];
    const float* bnm = (const float*)d_in[12]; const float* bnv = (const float*)d_in[13];
    const float* Wh  = (const float*)d_in[14]; const float* bh  = (const float*)d_in[15];
    (void)in_sizes; (void)n_in; (void)out_size; (void)ws_size;

    const int N = GN, E = GE;
    const int* src = ei;
    const int* dst = ei + E;

    char* p = (char*)d_ws;
    auto alloc = [&](size_t bytes) { char* r = p; p += (bytes + 255) & ~(size_t)255; return r; };
    float*    xcur  = (float*)   alloc((size_t)N * GHID * 4);
    _Float16* xh    = (_Float16*)alloc((size_t)N * GHID * 2);
    _Float16* qh    = (_Float16*)alloc((size_t)N * GHID * 2);
    _Float16* kh    = (_Float16*)alloc((size_t)N * GHID * 2);
    _Float16* vh    = (_Float16*)alloc((size_t)N * GHID * 2);
    float*    skip  = (float*)   alloc((size_t)N * GHID * 4);
    float*    alpha = (float*)   alloc((size_t)E * GH * 4);
    float*    amax  = (float*)   alloc((size_t)N * GH * 4);
    float*    denom = (float*)   alloc((size_t)N * GH * 4);
    _Float16* w16   = (_Float16*)alloc((size_t)3 * 4 * 4096 * 2);
    _Float16* wh16  = (_Float16*)alloc((size_t)GHID * GOC * 2);

    const int TB = 256;
    pack_weights<<<(3 * 4096 + TB - 1) / TB, TB, 0, stream>>>(Wq, Wk, Wv, Ws, Wh, w16, wh16);

    const float* xin = x;
    for (int l = 0; l < 3; ++l) {
        cvt_f32_to_f16<<<(N * GHID + TB - 1) / TB, TB, 0, stream>>>(xin, xh, N * GHID);
        gemm_qkvs<<<(N + 15) / 16, 128, 0, stream>>>(
            xh, w16 + (size_t)l * 4 * 4096,
            bq + l * GHID, bk + l * GHID, bv + l * GHID, bs + l * GHID,
            qh, kh, vh, skip, N);
        init_stats<<<(N * GH + TB - 1) / TB, TB, 0, stream>>>(amax, denom, N * GH);
        edge_logits<<<(E * GH + TB - 1) / TB, TB, 0, stream>>>(qh, kh, src, dst, alpha, amax, E);
        edge_exp<<<(E * GH + TB - 1) / TB, TB, 0, stream>>>(alpha, amax, dst, denom, E);
        edge_scatter<<<(E * GHID + TB - 1) / TB, TB, 0, stream>>>(vh, alpha, denom, src, dst, skip, E);
        bn_relu<<<(N * GHID + TB - 1) / TB, TB, 0, stream>>>(
            skip, bng + l * GHID, bnb + l * GHID, bnm + l * GHID, bnv + l * GHID, xcur, N * GHID);
        xin = xcur;
    }

    cvt_f32_to_f16<<<(N * GHID + TB - 1) / TB, TB, 0, stream>>>(xcur, xh, N * GHID);
    gemm_head<<<(N + 15) / 16, 64, 0, stream>>>(xh, wh16, bh, (float*)d_out, N);
}
